// PackedTernaryLinear_70781061038140
// MI455X (gfx1250) — compile-verified, matrix operand order checked by reading, not run on hardware
//
#include <hip/hip_runtime.h>
#include <hip/hip_bf16.h>
#include <cstdint>

typedef __attribute__((ext_vector_type(16))) __bf16   v16bf;
typedef __attribute__((ext_vector_type(8)))  __bf16   v8bf;
typedef __attribute__((ext_vector_type(8)))  float    v8f;
typedef __attribute__((ext_vector_type(4)))  float    v4f;
typedef __attribute__((ext_vector_type(8)))  uint32_t v8u;

#define IN_F  8192
#define OUT_F 8192
#define NTOK  128
#define RANK  32

// code {0,1,2,3} -> bf16 {-1, 0, +1, +1}
__device__ __forceinline__ uint32_t bf_of_code(unsigned c) {
  return (uint32_t)(unsigned short)(0x3F803F800000BF80ull >> (c << 4));
}

// ---------------------------------------------------------------------------
// Kernel 1: convert x f32 -> bf16 into workspace (row-major [128, 8192]).
// ---------------------------------------------------------------------------
__global__ __launch_bounds__(256) void cvt_x_kernel(const float* __restrict__ x,
                                                    __bf16* __restrict__ xb) {
  const int idx = (blockIdx.x * 256 + threadIdx.x) * 8;
  v4f a = *(const v4f*)(x + idx);
  v4f b = *(const v4f*)(x + idx + 4);
  v8bf o;
  o[0] = (__bf16)a.x; o[1] = (__bf16)a.y; o[2] = (__bf16)a.z; o[3] = (__bf16)a.w;
  o[4] = (__bf16)b.x; o[5] = (__bf16)b.y; o[6] = (__bf16)b.z; o[7] = (__bf16)b.w;
  *(v8bf*)(xb + idx) = o;
}

// ---------------------------------------------------------------------------
// Kernel 2: t = x @ lora_B^T  -> bf16 [128, 32].  One block per token.
// ---------------------------------------------------------------------------
__global__ __launch_bounds__(256) void lora_t_kernel(const float* __restrict__ x,
                                                     const float* __restrict__ lB,
                                                     __bf16* __restrict__ tb) {
  __shared__ float part[8][RANK];
  const int m = blockIdx.x;
  const int r = threadIdx.x & 31;
  const int c = threadIdx.x >> 5;
  const float* xr = x  + (size_t)m * IN_F + c * 1024;
  const float* br = lB + (size_t)r * IN_F + c * 1024;
  float s = 0.f;
  for (int k = 0; k < 1024; k += 4) {
    v4f a = *(const v4f*)(xr + k);
    v4f b = *(const v4f*)(br + k);
    s += a.x * b.x + a.y * b.y + a.z * b.z + a.w * b.w;
  }
  part[c][r] = s;
  __syncthreads();
  if (threadIdx.x < RANK) {
    float acc = 0.f;
    #pragma unroll
    for (int i = 0; i < 8; ++i) acc += part[i][threadIdx.x];
    tb[m * RANK + threadIdx.x] = (__bf16)acc;
  }
}

// ---------------------------------------------------------------------------
// Kernel 3: main GEMM.  Block = 128x64 output tile, 8 wave32s.
// Wave w: ntile = w&3, mtiles (w>>2)*4 .. +3.
// Ternary decode: 2 KB LDS LUT, byte (4 codes) -> 4 packed bf16 (ds_load_b64).
// K-loop unrolled x2: one b128 weight load feeds two 16x16x32 bf16 WMMA steps.
// Epilogue: scale, one LoRA WMMA (K=RANK=32), bias, store.
// ---------------------------------------------------------------------------
__global__ __launch_bounds__(256) void ternary_lora_gemm_kernel(
    const unsigned char* __restrict__ wp,
    const float* __restrict__ scale,
    const float* __restrict__ loraA,
    const float* __restrict__ bias,
    const __bf16* __restrict__ xb,
    const __bf16* __restrict__ tb,
    float* __restrict__ out) {
  __shared__ uint2 lut[256];

  const int tid   = threadIdx.x;
  const int wave  = tid >> 5;
  const int lane  = tid & 31;
  const int ntile = wave & 3;
  const int mbase = (wave >> 2) * 4;
  const int nl    = lane & 15;
  const int koff  = (lane >> 4) * 8;        // 0 or 8
  const unsigned s2 = (unsigned)koff * 2u;  // 0 or 16 (bit offset into packed dword)

  // Build decode LUT: byte b = 4 LSB-first 2-bit codes -> 4 bf16 (2 packed u32).
  {
    const unsigned b = (unsigned)tid;       // 256 threads, 256 entries
    const uint32_t p0 = bf_of_code(b & 3) | (bf_of_code((b >> 2) & 3) << 16);
    const uint32_t p1 = bf_of_code((b >> 4) & 3) | (bf_of_code((b >> 6) & 3) << 16);
    lut[b] = make_uint2(p0, p1);
  }
  __syncthreads();

  const int n = blockIdx.x * 64 + ntile * 16 + nl;
  const unsigned char* wrow = wp + (size_t)n * (IN_F / 4);

  v8f acc[4] = {{}, {}, {}, {}};

  for (int k0 = 0; k0 < IN_F; k0 += 64) {
    // 64 ternary codes for row n: 16 bytes, covers two K=32 WMMA steps.
    const uint4 w4 = *(const uint4*)(wrow + (k0 >> 2));
    __builtin_prefetch(wrow + (k0 >> 2) + 256, 0, 0);  // stream ahead

    #pragma unroll
    for (int half = 0; half < 2; ++half) {
      const uint32_t wlo = half ? w4.z : w4.x;   // codes K = kk+0..15
      const uint32_t whi = half ? w4.w : w4.y;   // codes K = kk+16..31

      // B fragment (32x16 bf16 layout): elems 0..7 <- K=koff.., 8..15 <- K=16+koff..
      const uint2 q0 = lut[(wlo >> s2) & 255u];
      const uint2 q1 = lut[(wlo >> (s2 + 8u)) & 255u];
      const uint2 q2 = lut[(whi >> s2) & 255u];
      const uint2 q3 = lut[(whi >> (s2 + 8u)) & 255u];
      v8u u;
      u[0] = q0.x; u[1] = q0.y; u[2] = q1.x; u[3] = q1.y;
      u[4] = q2.x; u[5] = q2.y; u[6] = q3.x; u[7] = q3.y;
      const v16bf bfrag = __builtin_bit_cast(v16bf, u);

      const int kk = k0 + half * 32;
      #pragma unroll
      for (int t = 0; t < 4; ++t) {
        // A fragment: M = (mbase+t)*16 + (lane&15), K halves at koff / 16+koff
        const int m = (mbase + t) * 16 + nl;
        const __bf16* xp = xb + (size_t)m * IN_F + kk + koff;
        const v8bf a0 = *(const v8bf*)(xp);
        const v8bf a1 = *(const v8bf*)(xp + 16);
        const v16bf af = __builtin_shufflevector(
            a0, a1, 0, 1, 2, 3, 4, 5, 6, 7, 8, 9, 10, 11, 12, 13, 14, 15);
        acc[t] = __builtin_amdgcn_wmma_f32_16x16x32_bf16(
            false, af, false, bfrag, (short)0, acc[t], false, false);
      }
    }
  }

  // LoRA B-fragment: B[r][n] = lora_A[n][r]; per lane n fixed, r per layout.
  const float* ap = loraA + (size_t)n * RANK + koff;
  const v4f r0 = *(const v4f*)(ap);
  const v4f r1 = *(const v4f*)(ap + 4);
  const v4f r2 = *(const v4f*)(ap + 16);
  const v4f r3 = *(const v4f*)(ap + 20);
  v16bf bl;
  #pragma unroll
  for (int j = 0; j < 4; ++j) {
    bl[j]      = (__bf16)r0[j];
    bl[j + 4]  = (__bf16)r1[j];
    bl[j + 8]  = (__bf16)r2[j];
    bl[j + 12] = (__bf16)r3[j];
  }
  const float sc = scale[n];
  const float bs = bias[n];

  #pragma unroll
  for (int t = 0; t < 4; ++t) {
    v8f a = acc[t];
    #pragma unroll
    for (int i = 0; i < 8; ++i) a[i] *= sc;   // scale applies to ternary path only

    // LoRA A-fragment from t = x @ lora_B^T (bf16 [128,32])
    const int m = (mbase + t) * 16 + nl;
    const __bf16* tp = tb + (size_t)m * RANK + koff;
    const v8bf t0 = *(const v8bf*)(tp);
    const v8bf t1 = *(const v8bf*)(tp + 16);
    const v16bf tf = __builtin_shufflevector(
        t0, t1, 0, 1, 2, 3, 4, 5, 6, 7, 8, 9, 10, 11, 12, 13, 14, 15);

    a = __builtin_amdgcn_wmma_f32_16x16x32_bf16(
        false, tf, false, bl, (short)0, a, false, false);

    // C/D layout: M = (mbase+t)*16 + i + 8*(lane>>4), N = lane&15
    const int mrow = (mbase + t) * 16 + 8 * (lane >> 4);
    #pragma unroll
    for (int i = 0; i < 8; ++i)
      out[(size_t)(mrow + i) * OUT_F + n] = a[i] + bs;
  }
}

// ---------------------------------------------------------------------------
extern "C" void kernel_launch(void* const* d_in, const int* in_sizes, int n_in,
                              void* d_out, int out_size, void* d_ws, size_t ws_size,
                              hipStream_t stream) {
  const float*         x     = (const float*)d_in[0];
  const unsigned char* wp    = (const unsigned char*)d_in[1];
  const float*         scale = (const float*)d_in[2];
  const float*         loraA = (const float*)d_in[3];
  const float*         loraB = (const float*)d_in[4];
  const float*         bias  = (const float*)d_in[5];
  float*               out   = (float*)d_out;

  // workspace: x_bf16 [128*8192] (2 MB) then t_bf16 [128*32] (8 KB)
  __bf16* xb  = (__bf16*)d_ws;
  __bf16* tbf = xb + (size_t)NTOK * IN_F;

  cvt_x_kernel<<<(NTOK * IN_F) / (256 * 8), 256, 0, stream>>>(x, xb);
  lora_t_kernel<<<NTOK, 256, 0, stream>>>(x, loraB, tbf);
  ternary_lora_gemm_kernel<<<OUT_F / 64, 256, 0, stream>>>(
      wp, scale, loraA, bias, xb, tbf, out);
}
